// QRNN_9337258901996
// MI455X (gfx1250) — matrix operand order repaired and checked by reading, not compiled, not run on hardware
//
#include <hip/hip_runtime.h>
#include <stdint.h>

// ---------------------------------------------------------------------------
// QRNN (window=2) for MI455X / gfx1250:
//   gates = x[t-1]@K0 + x[t]@K1 + bias  (bf16 WMMA, fp32 accum)
//   h_t   = o_t * (f_t * h_{t-1} + (1-f_t) * z_t)
// GEMM pipeline: double-buffered LDS staged with global_load_async_to_lds_b128
// (ASYNCcnt), one workgroup barrier per K-step, v_wmma_f32_16x16x32_bf16 core.
// ---------------------------------------------------------------------------

typedef __attribute__((ext_vector_type(16))) __bf16   v16bf;
typedef __attribute__((ext_vector_type(8)))  float    v8f;
typedef __attribute__((ext_vector_type(8)))  uint32_t v8u;

constexpr int kB  = 8;
constexpr int kT  = 1024;
constexpr int kD  = 1024;
constexpr int kU  = 1024;
constexpr int kN3 = 3 * kU;          // 3072
constexpr int kM  = kB * kT;         // 8192 GEMM rows
constexpr int kSteps = 32;           // 2 phases x (1024/64) stagings

// --- CDNA5 async global->LDS (signature probed in round 2) -----------------
#if __has_builtin(__builtin_amdgcn_global_load_async_to_lds_b128)
#define HAVE_ASYNC_LDS 1
#else
#define HAVE_ASYNC_LDS 0
#endif

#if HAVE_ASYNC_LDS
typedef __attribute__((__vector_size__(4 * sizeof(int)))) int b128_t;

__device__ __forceinline__ void async_load_b128(const void* g, void* l) {
  __builtin_amdgcn_global_load_async_to_lds_b128(
      (__attribute__((address_space(1))) b128_t*)(uintptr_t)g,
      (__attribute__((address_space(3))) b128_t*)l,
      0, 0);
}
__device__ __forceinline__ void wait_async0() {
#if __has_builtin(__builtin_amdgcn_s_wait_asynccnt)
  __builtin_amdgcn_s_wait_asynccnt(0);
#else
  asm volatile("s_wait_asynccnt 0x0" ::: "memory");
#endif
}
#else
__device__ __forceinline__ void wait_async0() {}
#endif

__device__ __forceinline__ uint16_t f32_bf16_rne(float f) {
  uint32_t u = __float_as_uint(f);
  u += 0x7FFFu + ((u >> 16) & 1u);   // round-to-nearest-even truncation
  return (uint16_t)(u >> 16);
}

// fp32 -> bf16, 4 elements per thread (float4 in, uint2 out)
__global__ void cvt_bf16_kernel(const float* __restrict__ in,
                                uint16_t* __restrict__ out, int n4) {
  int i = blockIdx.x * blockDim.x + threadIdx.x;
  if (i >= n4) return;
  float4 v = ((const float4*)in)[i];
  uint32_t lo = (uint32_t)f32_bf16_rne(v.x) | ((uint32_t)f32_bf16_rne(v.y) << 16);
  uint32_t hi = (uint32_t)f32_bf16_rne(v.z) | ((uint32_t)f32_bf16_rne(v.w) << 16);
  ((uint2*)out)[i] = make_uint2(lo, hi);
}

// Transpose-convert weights: in[w][d][n] f32 -> out[w][n][d] bf16 (32x32 tiles)
__global__ void __launch_bounds__(256)
cvt_kT_kernel(const float* __restrict__ in, uint16_t* __restrict__ out) {
  __shared__ float tile[32][33];
  const int w  = blockIdx.z;
  const int n0 = blockIdx.x * 32;
  const int d0 = blockIdx.y * 32;
  const int tx = threadIdx.x;         // 0..31
  const int ty = threadIdx.y;         // 0..7
  for (int j = 0; j < 32; j += 8)
    tile[ty + j][tx] = in[((size_t)w * kD + d0 + ty + j) * kN3 + n0 + tx];
  __syncthreads();
  for (int j = 0; j < 32; j += 8)
    out[((size_t)w * kN3 + n0 + ty + j) * kD + d0 + tx] =
        f32_bf16_rne(tile[tx][ty + j]);
}

// ---------------------------------------------------------------------------
// GEMM: gates[8192][3072] = x[t-1]@K0 + x[t]@K1, bf16 -> f32.
// Workgroup: 256 threads (8 wave32), 128x128 tile, K staged by 64,
// double-buffered async staging, 16 WMMAs per staging per wave.
// ---------------------------------------------------------------------------
__global__ void __launch_bounds__(256)
qrnn_gemm_kernel(const uint16_t* __restrict__ xbf,    // [B*T, D] bf16
                 const uint16_t* __restrict__ kbfT,   // [2, 3U, D] bf16 (pre-transposed)
                 float* __restrict__ gates) {         // [B*T, 3U] f32
  __shared__ __align__(16) uint16_t ldsA[2][128 * 64];  // [buf][row][k]  2x16 KB
  __shared__ __align__(16) uint16_t ldsB[2][128 * 64];  // [buf][n][k]    2x16 KB

  const int tid   = threadIdx.x;
  const int lane  = tid & 31;
  const int wave  = tid >> 5;
  const int half  = lane >> 4;        // lane-half select for CDNA5 frag layout
  const int l16   = lane & 15;
  const int waveM = wave >> 2;        // 0..1 -> 64-row slab
  const int waveN = wave & 3;         // 0..3 -> 32-col slab
  const int mBlk  = blockIdx.y;       // 0..63
  const int nBlk  = blockIdx.x;       // 0..23

  v8f acc[4][2];
  for (int mi = 0; mi < 4; ++mi)
    for (int nj = 0; nj < 2; ++nj)
      for (int e = 0; e < 8; ++e) acc[mi][nj][e] = 0.0f;

  // Per-thread staging slots: 4 x 16B segments; row = idx>>3, seg = idx&7.
  int rowS[4], segS[4];
  const uint16_t* aSrc0[4];           // x[m-1] base (w==0 phase)
  const uint16_t* bSrc0[4];           // K0^T base
  bool aOk0[4];                       // false only for the t==0 zero row
  for (int i = 0; i < 4; ++i) {
    int idx = tid + i * 256;
    rowS[i] = idx >> 3;
    segS[i] = idx & 7;
    int m = mBlk * 128 + rowS[i];
    aOk0[i] = (m & (kT - 1)) != 0;
    aSrc0[i] = xbf + (int64_t)(m - 1) * kD + segS[i] * 8;
    bSrc0[i] = kbfT + ((int64_t)nBlk * 128 + rowS[i]) * kD + segS[i] * 8;
  }

  // Stage step s (s = 0..31): w = s>>4 phase, K-offset (s&15)*64, into buffer.
  auto stage = [&](int s, int buf) {
    const int w  = s >> 4;
    const int ko = (s & 15) * 64;
    for (int i = 0; i < 4; ++i) {
      uint32_t loff = (uint32_t)(rowS[i] * 128 + segS[i] * 16);
      char* dA = (char*)ldsA[buf] + loff;
      char* dB = (char*)ldsB[buf] + loff;
      const uint16_t* pa = aSrc0[i] + (int64_t)w * kD + ko;           // w==1 -> x[m]
      const uint16_t* pb = bSrc0[i] + (int64_t)w * (kN3 * kD) + ko;   // w==1 -> K1^T
#if HAVE_ASYNC_LDS
      if (w || aOk0[i]) async_load_b128(pa, dA);
      else              *(uint4*)dA = make_uint4(0u, 0u, 0u, 0u);
      async_load_b128(pb, dB);
#else
      uint4 va = make_uint4(0u, 0u, 0u, 0u);
      if (w || aOk0[i]) va = *(const uint4*)pa;
      uint4 vb = *(const uint4*)pb;
      *(uint4*)dA = va;
      *(uint4*)dB = vb;
#endif
    }
  };

  auto compute = [&](int buf) {
    uint32_t* A32 = (uint32_t*)ldsA[buf];
    uint32_t* B32 = (uint32_t*)ldsB[buf];
    for (int c = 0; c < 2; ++c) {     // two K=32 chunks per staging
      // B fragments: VGPR p<4 -> K = half*8 + 2p; p>=4 -> 16 + half*8 + 2(p-4), N = l16.
      v8u bu[2];
      for (int nj = 0; nj < 2; ++nj) {
        int col = waveN * 32 + nj * 16 + l16;
        for (int p = 0; p < 8; ++p)
          bu[nj][p] = B32[col * 32 + c * 16 + (p >> 2) * 8 + half * 4 + (p & 3)];
      }
      // A fragments: VGPR p<4 -> K = half*8 + 2p; p>=4 -> 16 + half*8 + 2(p-4).
      for (int mi = 0; mi < 4; ++mi) {
        int row = waveM * 64 + mi * 16 + l16;
        v8u au;
        for (int p = 0; p < 4; ++p)
          au[p] = A32[row * 32 + c * 16 + half * 4 + p];
        for (int p = 0; p < 4; ++p)
          au[4 + p] = A32[row * 32 + c * 16 + 8 + half * 4 + p];
        v16bf afrag = __builtin_bit_cast(v16bf, au);
        for (int nj = 0; nj < 2; ++nj) {
          v16bf bfrag = __builtin_bit_cast(v16bf, bu[nj]);
          acc[mi][nj] = __builtin_amdgcn_wmma_f32_16x16x32_bf16(
              false, afrag, false, bfrag, (short)0, acc[mi][nj], false, false);
        }
      }
    }
  };

  // Software pipeline: prefetch batch s+1 into the other buffer while
  // computing batch s. The single barrier per step orders (a) batch-s
  // visibility (own asyncs waited to 0 before signaling; batch s+1 not yet
  // issued) and (b) reuse of the other buffer (frag ds_loads of step s-1 are
  // dscnt-waited before the signal).
  stage(0, 0);
  for (int s = 0; s < kSteps; ++s) {
    const int cur = s & 1;
    wait_async0();
    __syncthreads();
    if (s + 1 < kSteps) stage(s + 1, cur ^ 1);
    compute(cur);
  }

  // C/D layout: VGPR r -> M = r + 8*half; N = l16.
  for (int mi = 0; mi < 4; ++mi) {
    int rb = mBlk * 128 + waveM * 64 + mi * 16 + half * 8;
    for (int nj = 0; nj < 2; ++nj) {
      int col = nBlk * 128 + waveN * 32 + nj * 16 + l16;
      for (int r = 0; r < 8; ++r)
        gates[(size_t)(rb + r) * kN3 + col] = acc[mi][nj][r];
    }
  }
}

// ---------------------------------------------------------------------------
// Sequential scan: one thread per (b,u) channel; coalesced gate reads across u.
// ---------------------------------------------------------------------------
__global__ void __launch_bounds__(256)
qrnn_scan_kernel(const float* __restrict__ gates,   // [B*T, 3U]
                 const float* __restrict__ bias,    // [3U]
                 float* __restrict__ out) {         // [B, T, U]
  int tid = blockIdx.x * blockDim.x + threadIdx.x;  // 0..8191
  int b = tid >> 10;
  int u = tid & (kU - 1);
  float b0 = bias[u];
  float b1 = bias[kU + u];
  float b2 = bias[2 * kU + u];
  const float* g = gates + (size_t)b * kT * kN3;
  float* o = out + (size_t)b * kT * kU + u;
  float h = 0.0f;
  for (int t = 0; t < kT; ++t) {
    const float* gt = g + (size_t)t * kN3;
    float z  = tanhf(gt[u] + b0);
    float f  = 1.0f / (1.0f + __expf(-(gt[kU + u] + b1)));
    float oo = 1.0f / (1.0f + __expf(-(gt[2 * kU + u] + b2)));
    h = oo * (f * h + (1.0f - f) * z);
    o[(size_t)t * kU] = h;
  }
}

// ---------------------------------------------------------------------------
extern "C" void kernel_launch(void* const* d_in, const int* in_sizes, int n_in,
                              void* d_out, int out_size, void* d_ws, size_t ws_size,
                              hipStream_t stream) {
  const float* x    = (const float*)d_in[0];  // [8,1024,1024]
  const float* kern = (const float*)d_in[1];  // [2,1024,3072]
  const float* bias = (const float*)d_in[2];  // [3072]
  float* out = (float*)d_out;                 // [8,1024,1024]

  const size_t nX = (size_t)kM * kD;          // 8388608 elems
  const size_t nK = (size_t)2 * kD * kN3;     // 6291456 elems
  // ws layout: x_bf16 (16 MB) | kT_bf16 (12 MB) | gates f32 (96 MB) ~= 130 MB
  uint16_t* xbf   = (uint16_t*)d_ws;
  uint16_t* kbfT  = xbf + nX;
  float*    gates = (float*)(kbfT + nK);

  cvt_bf16_kernel<<<(int)((nX / 4 + 255) / 256), 256, 0, stream>>>(x, xbf, (int)(nX / 4));
  cvt_kT_kernel<<<dim3(kN3 / 32, kD / 32, 2), dim3(32, 8), 0, stream>>>(kern, kbfT);

  qrnn_gemm_kernel<<<dim3(kN3 / 128, kM / 128), 256, 0, stream>>>(xbf, kbfT, gates);

  qrnn_scan_kernel<<<(kB * kU) / 256, 256, 0, stream>>>(gates, bias, out);
}
